// MixedGRUCritic_15564961481122
// MI455X (gfx1250) — matrix operand order repaired and checked by reading, not compiled, not run on hardware
//
#include <hip/hip_runtime.h>
#include <hip/hip_bf16.h>
#include <math.h>

typedef __attribute__((ext_vector_type(16))) __bf16 v16bf;
typedef __attribute__((ext_vector_type(8)))  __bf16 v8bf;
typedef __attribute__((ext_vector_type(8)))  float  v8f;

#define Hd     256
#define G3     768
#define Bsz    32
#define Ndim   64
#define Mdim   64
#define Cdim   64
#define C1d    16
#define INTER  912
#define INTERP 928
#define FW     912
#define GW     (3*INTER + Hd)   // 2992

#define USE_ASYNC_LDS 1

// ---------- helpers ----------

__device__ __forceinline__ __bf16 f2bf(float f) {
  union { float f; unsigned u; } v; v.f = f;
  unsigned r = v.u + 0x7FFFu + ((v.u >> 16) & 1u);   // RNE
  return __builtin_bit_cast(__bf16, (unsigned short)(r >> 16));
}
__device__ __forceinline__ __bf16 bf_zero() {
  return __builtin_bit_cast(__bf16, (unsigned short)0);
}
__device__ __forceinline__ float sigmoidf(float x) { return 1.f / (1.f + expf(-x)); }

// A-matrix 16x32 bf16 fragment from LDS (ISA 7.12.2):
// lane<16: K = kb..kb+15; lane>=16: K = kb+16..kb+31 (matches B)
__device__ __forceinline__ v16bf loadA_lds(const __bf16* base, int ld, int kbase, int lane) {
  const __bf16* p = base + (lane & 15) * ld + kbase + (lane >> 4) * 16;
  union { v16bf v; v8bf h[2]; } u;
  u.h[0] = *(const v8bf*)(p);
  u.h[1] = *(const v8bf*)(p + 8);
  return u.v;
}

// B-matrix 32x16 bf16 fragment: B[k,n]=W[ncol+n,k], per-lane 16 contiguous K.
// lanes 0-15: K=kb..kb+15 (col=lane), lanes 16-31: K=kb+16..kb+31
__device__ __forceinline__ v16bf loadB_glb(const __bf16* W, int ldw, int ncol, int kbase, int lane) {
  return *(const v16bf*)(W + (size_t)(ncol + (lane & 15)) * ldw + kbase + (lane >> 4) * 16);
}
__device__ __forceinline__ v16bf loadB_lds(const __bf16* W, int ldw, int ncol, int kbase, int lane) {
  const __bf16* p = W + (ncol + (lane & 15)) * ldw + kbase + (lane >> 4) * 16;
  union { v16bf v; v8bf h[2]; } u;
  u.h[0] = *(const v8bf*)(p);
  u.h[1] = *(const v8bf*)(p + 8);
  return u.v;
}

#define WMMA_BF16(A, B, C) \
  __builtin_amdgcn_wmma_f32_16x16x32_bf16(false, (A), false, (B), (short)0, (C), false, false)

#if USE_ASYNC_LDS
__device__ __forceinline__ void async_cp16(const __bf16* gsrc, __bf16* ldst) {
  unsigned long long ga = (unsigned long long)(uintptr_t)gsrc;
  unsigned ldsOff = (unsigned)(uintptr_t)ldst;
  asm volatile("global_load_async_to_lds_b128 %0, %1, off" :: "v"(ldsOff), "v"(ga) : "memory");
}
#endif

// ---------- fp32 -> bf16 weight conversion (zero K-padding) ----------

__global__ void cvt_kernel(const float* __restrict__ src, __bf16* __restrict__ dst,
                           int rows, int cols, int ldo) {
  int idx = blockIdx.x * blockDim.x + threadIdx.x;
  if (idx >= rows * ldo) return;
  int r = idx / ldo, c = idx - r * ldo;
  dst[idx] = (c < cols) ? f2bf(src[r * cols + c]) : bf_zero();
}

__global__ void cvt_states_kernel(const float* __restrict__ src, __bf16* __restrict__ dst, int n) {
  int idx = blockIdx.x * blockDim.x + threadIdx.x;
  if (idx < n) dst[idx] = f2bf(src[idx]);
}

// ---------- six pooling reductions + x1 concat into f ----------

__global__ void pool_kernel(const float* __restrict__ states, const float* __restrict__ x1,
                            float* __restrict__ f) {
  int idx = blockIdx.x * blockDim.x + threadIdx.x;
  if (idx >= Bsz * 64 * Cdim) return;
  int c = idx & 63, j = (idx >> 6) & 63, b = idx >> 12;
  const float* sb = states + (size_t)b * Ndim * Mdim * Cdim;
  float mx1 = -INFINITY, mn1 = INFINITY, s1 = 0.f;
  float mx2 = -INFINITY, mn2 = INFINITY, s2 = 0.f;
  for (int n = 0; n < Ndim; n++) {
    float v = sb[((size_t)n * Mdim + j) * Cdim + c];
    mx1 = fmaxf(mx1, v); mn1 = fminf(mn1, v); s1 += v;
  }
  for (int m = 0; m < Mdim; m++) {
    float v = sb[((size_t)j * Mdim + m) * Cdim + c];
    mx2 = fmaxf(mx2, v); mn2 = fminf(mn2, v); s2 += v;
  }
  float* fr = f + (size_t)(b * 64 + j) * FW;
  fr[ 16 + c] = mx1;
  fr[ 80 + c] = s1 * (1.f / 64.f);
  fr[144 + c] = mn1;
  fr[208 + c] = mx2;
  fr[272 + c] = s2 * (1.f / 64.f);
  fr[336 + c] = mn2;
  if (c < C1d) fr[c] = x1[(size_t)(b * 64 + j) * C1d + c];
}

// ---------- gru2d recurrence (sd1 / sd2 via blockIdx.y) ----------
// 16 seq/block, 16 waves; wave w owns cols [w*16, w*16+16) of each gate.
// Whh r/z fragments register-resident (128 VGPRs); Wih + Whh_n staged in LDS.

__global__ __launch_bounds__(512, 1) void gru2d_kernel(
    const __bf16* __restrict__ statesB, const int* __restrict__ perm1, const int* __restrict__ perm2,
    const __bf16* __restrict__ WihB, const __bf16* __restrict__ WhhB,
    const float* __restrict__ bih, const float* __restrict__ bhh,
    float* __restrict__ f) {
  __shared__ __align__(32) __bf16 WihL[G3 * Cdim];    //  96KB: Wih, all 768 cols
  __shared__ __align__(32) __bf16 WhhLn[Hd * Hd];     // 128KB: n-gate Whh (cols 512..767)
  __shared__ __align__(32) __bf16 xB[16 * Cdim];      //   2KB
  __shared__ __align__(32) float  hF[16 * Hd];        //  16KB
  __shared__ __align__(32) __bf16 hB[16 * Hd];        //   8KB

  const int tid = threadIdx.x, lane = tid & 31, w = tid >> 5;
  const int mode = blockIdx.y;
  const int seqBase = blockIdx.x * 16;

  for (int i = tid; i < 16 * Hd; i += 512) { hF[i] = 0.f; hB[i] = bf_zero(); }

  // one-time async preload of LDS-resident weights
#if USE_ASYNC_LDS
  for (int i = tid; i < (G3 * Cdim) / 8; i += 512) async_cp16(WihB + i * 8, &WihL[i * 8]);
  for (int i = tid; i < (Hd * Hd) / 8; i += 512)   async_cp16(WhhB + 2 * Hd * Hd + i * 8, &WhhLn[i * 8]);
  asm volatile("s_wait_asynccnt 0x0" ::: "memory");
#else
  for (int i = tid; i < G3 * Cdim; i += 512) WihL[i] = WihB[i];
  for (int i = tid; i < Hd * Hd; i += 512)   WhhLn[i] = WhhB[2 * Hd * Hd + i];
#endif

  // register-resident r/z recurrent weights: 2 gates x 8 K-chunks
  v16bf BWrz[2][8];
  #pragma unroll
  for (int g = 0; g < 2; g++)
    #pragma unroll
    for (int kc = 0; kc < 8; kc++)
      BWrz[g][kc] = loadB_glb(WhhB, Hd, g * Hd + w * 16, kc * 32, lane);

  const int colL = w * 16 + (lane & 15);
  const float bR  = bih[colL]          + bhh[colL];
  const float bZ  = bih[Hd + colL]     + bhh[Hd + colL];
  const float bNI = bih[2 * Hd + colL];
  const float bNH = bhh[2 * Hd + colL];
  const int rowBase = (lane >> 4) * 8;
  __syncthreads();

  for (int tstep = 0; tstep < Mdim; tstep++) {
    // stage gathered x_t tile (16 rows x 128B) via async copy
#if USE_ASYNC_LDS
    if (tid < 128) {
      int r = tid >> 3, seg = tid & 7;
      int seq = seqBase + r, b = seq >> 6, n = seq & 63;
      size_t rowOff;
      if (mode == 0) {
        int m = perm1[(n << 6) + tstep];
        rowOff = (((size_t)(b * Ndim + n)) * Mdim + m) * Cdim;
      } else {
        int nn = perm2[(n << 6) + tstep];
        rowOff = (((size_t)(b * Ndim + nn)) * Mdim + n) * Cdim;
      }
      async_cp16(statesB + rowOff + seg * 8, &xB[r * Cdim + seg * 8]);
    }
    asm volatile("s_wait_asynccnt 0x0" ::: "memory");
#else
    for (int i = tid; i < 16 * Cdim; i += 512) {
      int r = i >> 6, c = i & 63;
      int seq = seqBase + r, b = seq >> 6, n = seq & 63;
      size_t src;
      if (mode == 0) {
        int m = perm1[(n << 6) + tstep];
        src = (((size_t)(b * Ndim + n)) * Mdim + m) * Cdim + c;
      } else {
        int nn = perm2[(n << 6) + tstep];
        src = (((size_t)(b * Ndim + nn)) * Mdim + n) * Cdim + c;
      }
      xB[i] = statesB[src];
    }
#endif
    __syncthreads();

    v8f aR, aZ, aNI, aNH;
    #pragma unroll
    for (int j = 0; j < 8; j++) { aR[j] = bR; aZ[j] = bZ; aNI[j] = bNI; aNH[j] = bNH; }

    #pragma unroll
    for (int kc = 0; kc < 2; kc++) {                  // gi (K=64), B from LDS
      v16bf a = loadA_lds(xB, Cdim, kc * 32, lane);
      aR  = WMMA_BF16(a, loadB_lds(WihL, Cdim, 0 * Hd + w * 16, kc * 32, lane), aR);
      aZ  = WMMA_BF16(a, loadB_lds(WihL, Cdim, 1 * Hd + w * 16, kc * 32, lane), aZ);
      aNI = WMMA_BF16(a, loadB_lds(WihL, Cdim, 2 * Hd + w * 16, kc * 32, lane), aNI);
    }
    #pragma unroll
    for (int kc = 0; kc < 8; kc++) {                  // gh (K=256): r/z resident, n from LDS
      v16bf a = loadA_lds(hB, Hd, kc * 32, lane);
      aR  = WMMA_BF16(a, BWrz[0][kc], aR);
      aZ  = WMMA_BF16(a, BWrz[1][kc], aZ);
      aNH = WMMA_BF16(a, loadB_lds(WhhLn, Hd, w * 16, kc * 32, lane), aNH);
    }
    __syncthreads();                                  // all hB/xB reads done

    #pragma unroll
    for (int j = 0; j < 8; j++) {
      float r = sigmoidf(aR[j]);
      float z = sigmoidf(aZ[j]);
      float n = tanhf(aNI[j] + r * aNH[j]);
      int off = (rowBase + j) * Hd + colL;
      float hn = (1.f - z) * n + z * hF[off];
      hF[off] = hn;
      hB[off] = f2bf(hn);
    }
    __syncthreads();
  }

  const int fOff = C1d + 6 * Cdim + (mode ? Hd : 0);  // 400 / 656
  for (int i = tid; i < 16 * Hd; i += 512) {
    int r = i >> 8, c = i & 255;
    int seq = seqBase + r, b = seq >> 6, n = seq & 63;
    f[((size_t)(b * Ndim + n)) * FW + fOff + c] = hF[i];
  }
}

// ---------- precompute gru1d input projection xW1 = f @ Wih1^T + bih1 ----------
// Fully parallel; output layout [t][col][batch] for contiguous recurrent reads.

__global__ __launch_bounds__(512, 1) void xw1_kernel(
    const float* __restrict__ f, const __bf16* __restrict__ WihB,
    const float* __restrict__ bih, float* __restrict__ xW1T) {
  __shared__ __align__(32) __bf16 xB[16 * INTERP];
  const int tid = threadIdx.x, lane = tid & 31, w = tid >> 5;
  const int t = blockIdx.x, bt = blockIdx.y;

  for (int i = tid; i < 16 * INTER; i += 512) {
    int r = i / INTER, c = i - r * INTER;
    int b = bt * 16 + r;
    xB[r * INTERP + c] = f2bf(f[((size_t)(b * Ndim + t)) * FW + c]);
  }
  for (int i = tid; i < 16 * (INTERP - INTER); i += 512) {
    int r = i >> 4, c = i & 15;
    xB[r * INTERP + INTER + c] = bf_zero();
  }
  __syncthreads();

  v8f acc[3];
  int colL[3];
  #pragma unroll
  for (int s = 0; s < 3; s++) {
    colL[s] = s * Hd + w * 16 + (lane & 15);
    float bv = bih[colL[s]];
    #pragma unroll
    for (int j = 0; j < 8; j++) acc[s][j] = bv;
  }
  for (int kc = 0; kc < INTERP; kc += 32) {           // 29 K-chunks, B streamed
    v16bf a = loadA_lds(xB, INTERP, kc, lane);
    #pragma unroll
    for (int s = 0; s < 3; s++)
      acc[s] = WMMA_BF16(a, loadB_glb(WihB, INTERP, s * Hd + w * 16, kc, lane), acc[s]);
  }
  const int rowBase = (lane >> 4) * 8;
  #pragma unroll
  for (int s = 0; s < 3; s++)
    #pragma unroll
    for (int j = 0; j < 8; j++)
      xW1T[((size_t)(t * G3 + colL[s])) * Bsz + bt * 16 + rowBase + j] = acc[s][j];
}

// ---------- gru1d recurrence (h part only; gi precomputed) ----------

__global__ __launch_bounds__(512, 1) void gru1d_kernel(
    const float* __restrict__ xW1T, const __bf16* __restrict__ WhhB,
    const float* __restrict__ bhh, float* __restrict__ hlast) {
  __shared__ __align__(32) __bf16 WhhLn[Hd * Hd];     // 128KB: n-gate Whh1
  __shared__ __align__(32) float  hF[16 * Hd];
  __shared__ __align__(32) __bf16 hB[16 * Hd];
  const int tid = threadIdx.x, lane = tid & 31, w = tid >> 5;
  const int bt = blockIdx.x;

  for (int i = tid; i < 16 * Hd; i += 512) { hF[i] = 0.f; hB[i] = bf_zero(); }
#if USE_ASYNC_LDS
  for (int i = tid; i < (Hd * Hd) / 8; i += 512) async_cp16(WhhB + 2 * Hd * Hd + i * 8, &WhhLn[i * 8]);
  asm volatile("s_wait_asynccnt 0x0" ::: "memory");
#else
  for (int i = tid; i < Hd * Hd; i += 512) WhhLn[i] = WhhB[2 * Hd * Hd + i];
#endif

  v16bf BWrz[2][8];
  #pragma unroll
  for (int g = 0; g < 2; g++)
    #pragma unroll
    for (int kc = 0; kc < 8; kc++)
      BWrz[g][kc] = loadB_glb(WhhB, Hd, g * Hd + w * 16, kc * 32, lane);

  const int colL = w * 16 + (lane & 15);
  const float bhr = bhh[colL], bhz = bhh[Hd + colL], bhn = bhh[2 * Hd + colL];
  const int rowBase = (lane >> 4) * 8;
  __syncthreads();

  for (int tstep = 0; tstep < Ndim; tstep++) {
    const float* gi = xW1T + (size_t)tstep * G3 * Bsz + bt * 16 + rowBase;
    v8f giR = *(const v8f*)(gi + (size_t)colL * Bsz);
    v8f giZ = *(const v8f*)(gi + (size_t)(Hd + colL) * Bsz);
    v8f giN = *(const v8f*)(gi + (size_t)(2 * Hd + colL) * Bsz);

    v8f aR, aZ, aNH;
    #pragma unroll
    for (int j = 0; j < 8; j++) { aR[j] = giR[j] + bhr; aZ[j] = giZ[j] + bhz; aNH[j] = bhn; }

    #pragma unroll
    for (int kc = 0; kc < 8; kc++) {
      v16bf a = loadA_lds(hB, Hd, kc * 32, lane);
      aR  = WMMA_BF16(a, BWrz[0][kc], aR);
      aZ  = WMMA_BF16(a, BWrz[1][kc], aZ);
      aNH = WMMA_BF16(a, loadB_lds(WhhLn, Hd, w * 16, kc * 32, lane), aNH);
    }
    __syncthreads();

    #pragma unroll
    for (int j = 0; j < 8; j++) {
      float r = sigmoidf(aR[j]);
      float z = sigmoidf(aZ[j]);
      float n = tanhf(giN[j] + r * aNH[j]);
      int off = (rowBase + j) * Hd + colL;
      float hn = (1.f - z) * n + z * hF[off];
      hF[off] = hn;
      hB[off] = f2bf(hn);
    }
    __syncthreads();
  }

  for (int i = tid; i < 16 * Hd; i += 512) {
    int r = i >> 8, c = i & 255;
    hlast[(size_t)(bt * 16 + r) * Hd + c] = hF[i];
  }
}

// ---------- pooling over N + fc1(relu) + fc2 ----------

__global__ __launch_bounds__(256) void head_kernel(
    const float* __restrict__ f, const float* __restrict__ hlast,
    const float* __restrict__ fc1W, const float* __restrict__ fc1b,
    const float* __restrict__ fc2W, const float* __restrict__ fc2b,
    float* __restrict__ out) {
  __shared__ float g[GW];
  __shared__ float red[256];
  const int b = blockIdx.x, tid = threadIdx.x;

  for (int j = tid; j < INTER; j += 256) {
    float mx = -INFINITY, mn = INFINITY, s = 0.f;
    const float* fb = f + (size_t)b * Ndim * FW + j;
    for (int n = 0; n < Ndim; n++) {
      float v = fb[(size_t)n * FW];
      mx = fmaxf(mx, v); mn = fminf(mn, v); s += v;
    }
    g[j] = mx; g[INTER + j] = s * (1.f / 64.f); g[2 * INTER + j] = mn;
  }
  g[3 * INTER + tid] = hlast[b * Hd + tid];
  __syncthreads();

  float o = fc1b[tid];
  const float* wr = fc1W + (size_t)tid * GW;
  for (int k = 0; k < GW; k++) o += g[k] * wr[k];
  red[tid] = fmaxf(o, 0.f) * fc2W[tid];
  __syncthreads();
  for (int s = 128; s > 0; s >>= 1) {
    if (tid < s) red[tid] += red[tid + s];
    __syncthreads();
  }
  if (tid == 0) out[b] = red[0] + fc2b[0];
}

// ---------- host launch ----------

extern "C" void kernel_launch(void* const* d_in, const int* in_sizes, int n_in,
                              void* d_out, int out_size, void* d_ws, size_t ws_size,
                              hipStream_t stream) {
  const float* x1     = (const float*)d_in[0];
  const float* states = (const float*)d_in[1];
  const int*   perm1  = (const int*)d_in[2];
  const int*   perm2  = (const int*)d_in[3];
  const float* g2_Wih = (const float*)d_in[4];
  const float* g2_Whh = (const float*)d_in[5];
  const float* g2_bih = (const float*)d_in[6];
  const float* g2_bhh = (const float*)d_in[7];
  const float* g1_Wih = (const float*)d_in[8];
  const float* g1_Whh = (const float*)d_in[9];
  const float* g1_bih = (const float*)d_in[10];
  const float* g1_bhh = (const float*)d_in[11];
  const float* fc1W   = (const float*)d_in[12];
  const float* fc1b   = (const float*)d_in[13];
  const float* fc2W   = (const float*)d_in[14];
  const float* fc2b   = (const float*)d_in[15];
  float* out = (float*)d_out;

  char* ws = (char*)d_ws;
  __bf16* Wih2B   = (__bf16*)(ws + 0);          //  768x64  bf16   ->   98304
  __bf16* Whh2B   = (__bf16*)(ws + 98304);      //  768x256 bf16   ->  491520
  __bf16* Wih1B   = (__bf16*)(ws + 491520);     //  768x928 bf16   -> 1916928
  __bf16* Whh1B   = (__bf16*)(ws + 1916928);    //  768x256 bf16   -> 2310144
  __bf16* statesB = (__bf16*)(ws + 2310144);    //  32x64x64x64 bf16 -> 19087360
  float*  f       = (float*)(ws + 19087360);    //  32x64x912 f32  -> 26558464
  float*  xW1T    = (float*)(ws + 26558464);    //  64x768x32 f32  -> 32849920
  float*  hlast   = (float*)(ws + 32849920);    //  32x256 f32     -> 32882688

  const int nStates = Bsz * Ndim * Mdim * Cdim;

  cvt_kernel<<<(G3 * Cdim   + 255) / 256, 256, 0, stream>>>(g2_Wih, Wih2B, G3, Cdim,  Cdim);
  cvt_kernel<<<(G3 * Hd     + 255) / 256, 256, 0, stream>>>(g2_Whh, Whh2B, G3, Hd,    Hd);
  cvt_kernel<<<(G3 * INTERP + 255) / 256, 256, 0, stream>>>(g1_Wih, Wih1B, G3, INTER, INTERP);
  cvt_kernel<<<(G3 * Hd     + 255) / 256, 256, 0, stream>>>(g1_Whh, Whh1B, G3, Hd,    Hd);
  cvt_states_kernel<<<(nStates + 255) / 256, 256, 0, stream>>>(states, statesB, nStates);

  pool_kernel<<<(Bsz * 64 * Cdim + 255) / 256, 256, 0, stream>>>(states, x1, f);

  gru2d_kernel<<<dim3(128, 2), 512, 0, stream>>>(statesB, perm1, perm2,
                                                 Wih2B, Whh2B, g2_bih, g2_bhh, f);

  xw1_kernel<<<dim3(Ndim, 2), 512, 0, stream>>>(f, Wih1B, g1_bih, xW1T);

  gru1d_kernel<<<2, 512, 0, stream>>>(xW1T, Whh1B, g1_bhh, hlast);

  head_kernel<<<Bsz, 256, 0, stream>>>(f, hlast, fc1W, fc1b, fc2W, fc2b, out);
}
